// GCN_Net_4209067950741
// MI455X (gfx1250) — matrix operand order repaired and checked by reading, not compiled, not run on hardware
//
#include <hip/hip_runtime.h>

typedef __attribute__((ext_vector_type(16))) _Float16 v16h;
typedef __attribute__((ext_vector_type(8)))  _Float16 v8h;
typedef __attribute__((ext_vector_type(4)))  _Float16 v4h;
typedef __attribute__((ext_vector_type(8)))  float    v8f;
typedef __attribute__((ext_vector_type(4)))  float    v4f;

#define GCN_F_IN 512
#define GCN_HID  64
#define GCN_NCLS 40

// ------------------------------ small kernels ------------------------------

__global__ void k_fill_ones(float* p, int n) {
    int i = blockIdx.x * blockDim.x + threadIdx.x;
    if (i < n) p[i] = 1.0f;
}

__global__ void k_deg_accum(const int* __restrict__ cols, float* __restrict__ deg, int E) {
    int e = blockIdx.x * blockDim.x + threadIdx.x;
    if (e < E) atomicAdd(&deg[cols[e]], 1.0f);
}

__global__ void k_inv_sqrt(float* p, int n) {
    int i = blockIdx.x * blockDim.x + threadIdx.x;
    if (i < n) { float d = p[i]; p[i] = (d > 0.0f) ? rsqrtf(d) : 0.0f; }
}

// W1 [512][64] f32 -> W1t [64][512] f16
__global__ void k_w1_to_half_t(const float* __restrict__ W, _Float16* __restrict__ Wt) {
    int i = blockIdx.x * blockDim.x + threadIdx.x;
    if (i >= GCN_HID * GCN_F_IN) return;
    int n = i / GCN_F_IN, k = i % GCN_F_IN;
    Wt[i] = (_Float16)W[k * GCN_HID + n];
}

// W2 [64][40] f32 -> W2t [64][64] f16, zero-padded cols 40..63
__global__ void k_w2_to_half_t(const float* __restrict__ W, _Float16* __restrict__ Wt) {
    int i = blockIdx.x * blockDim.x + threadIdx.x;
    if (i >= 64 * 64) return;
    int n = i / 64, k = i % 64;
    Wt[i] = (n < GCN_NCLS) ? (_Float16)W[k * GCN_NCLS + n] : (_Float16)0.0f;
}

// acc[col[e]][f] += dinv[row]*dinv[col] * h[row[e]][f]
template <int F>
__global__ void k_agg_edges(const int* __restrict__ rows, const int* __restrict__ cols,
                            const float* __restrict__ dinv, const float* __restrict__ h,
                            float* __restrict__ acc, int E) {
    long long t = (long long)blockIdx.x * blockDim.x + threadIdx.x;
    if (t >= (long long)E * F) return;
    int e = (int)(t / F), f = (int)(t % F);
    int r = rows[e], c = cols[e];
    float w = dinv[r] * dinv[c];
    atomicAdd(&acc[(long long)c * F + f], w * h[(long long)r * F + f]);
}

__global__ void k_relu_ip(float* p, long long n) {
    long long i = (long long)blockIdx.x * blockDim.x + threadIdx.x;
    if (i < n) p[i] = fmaxf(p[i], 0.0f);
}

// one wave per row; ncols <= 64
__global__ __launch_bounds__(128)
void k_log_softmax_rows(float* __restrict__ out, int nrows, int ncols) {
    int wave = threadIdx.x >> 5, lane = threadIdx.x & 31;
    int row = blockIdx.x * 4 + wave;
    if (row >= nrows) return;
    float* p = out + (long long)row * ncols;
    float neginf = -__builtin_inff();
    float v0 = (lane < ncols) ? p[lane] : neginf;
    float v1 = (lane + 32 < ncols) ? p[lane + 32] : neginf;
    float m = fmaxf(v0, v1);
    #pragma unroll
    for (int s = 16; s > 0; s >>= 1) m = fmaxf(m, __shfl_xor(m, s, 32));
    float sum = ((lane < ncols) ? __expf(v0 - m) : 0.0f) +
                ((lane + 32 < ncols) ? __expf(v1 - m) : 0.0f);
    #pragma unroll
    for (int s = 16; s > 0; s >>= 1) sum += __shfl_xor(sum, s, 32);
    float lse = m + __logf(sum);
    if (lane < ncols) p[lane] = v0 - lse;
    if (lane + 32 < ncols) p[lane + 32] = v1 - lse;
}

// --------------------- WMMA GEMM: C = A(f32) * B(f16^T) ---------------------
// Block: 4 waves, 16 rows x 64 cols of C; wave w owns 16x16 tile at col w*16.
// Bt is [Npad][K] f16 (row n = column n of B). Double-buffered LDS A staging
// (one barrier per K-step) + register prefetch of the next A tile issued after
// the fragment loads so its latency overlaps the WMMA.
// Fused epilogue: C[r][n] = val, Cinit[r][n] = dinv[r]^2 * val + bias[n].
template <int K, int LDA, int NOUT, int LDC>
__global__ __launch_bounds__(128)
void k_gemm_wmma(const float* __restrict__ A, const _Float16* __restrict__ Bt,
                 float* __restrict__ C, float* __restrict__ Cinit,
                 const float* __restrict__ dinv, const float* __restrict__ bias,
                 int nrows) {
    constexpr int NK = K / 32;
    __shared__ __align__(16) _Float16 lds_a[2][16][32];
    const int tid   = threadIdx.x;
    const int wave  = tid >> 5;
    const int lane  = tid & 31;
    const int row0  = blockIdx.x * 16;
    const int n0    = wave * 16;
    const int m     = lane & 15;
    const int hoff  = (lane < 16) ? 0 : 8;    // A fragment: K-half offset per ISA layout
    const int koffB = (lane < 16) ? 0 : 8;    // B fragment: K offset per ISA layout
    const int ldr   = tid >> 3;               // staging row 0..15
    const int kq    = (tid & 7) << 2;         // staging K quad 0,4,..,28
    const int arow  = min(row0 + ldr, nrows - 1);
    const float* arp = &A[(long long)arow * LDA + kq];

    // preload first A tile into registers
    v4f xa = *reinterpret_cast<const v4f*>(arp);

    v8f acc = {};
    #pragma unroll
    for (int kk = 0; kk < NK; ++kk) {
        const int k0 = kk * 32;
        // stage current tile (f32 -> f16) into LDS buffer kk&1
        v4h ha;
        ha[0] = (_Float16)xa[0]; ha[1] = (_Float16)xa[1];
        ha[2] = (_Float16)xa[2]; ha[3] = (_Float16)xa[3];
        *reinterpret_cast<v4h*>(&lds_a[kk & 1][ldr][kq]) = ha;
        __syncthreads();

        // fragment loads for this step
        v8h alo = *reinterpret_cast<const v8h*>(&lds_a[kk & 1][m][hoff]);
        v8h ahi = *reinterpret_cast<const v8h*>(&lds_a[kk & 1][m][16 + hoff]);
        const _Float16* br = &Bt[(long long)(n0 + m) * K + k0 + koffB];
        v8h blo = *reinterpret_cast<const v8h*>(br);
        v8h bhi = *reinterpret_cast<const v8h*>(br + 16);

        // prefetch next A tile (issued after B loads -> overlaps the WMMA)
        if (kk + 1 < NK)
            xa = *reinterpret_cast<const v4f*>(arp + k0 + 32);

        v16h a, b;
        #pragma unroll
        for (int j = 0; j < 8; ++j) {
            a[j] = alo[j]; a[j + 8] = ahi[j];
            b[j] = blo[j]; b[j + 8] = bhi[j];
        }
        acc = __builtin_amdgcn_wmma_f32_16x16x32_f16(false, a, false, b,
                                                     (short)0, acc, false, false);
    }

    const int n = n0 + m;
    if (n < NOUT) {
        const float bn = bias[n];
        const int rbase = row0 + ((lane < 16) ? 0 : 8);
        #pragma unroll
        for (int v = 0; v < 8; ++v) {
            int r = rbase + v;
            if (r < nrows) {
                float val = acc[v];
                C[(long long)r * LDC + n] = val;
                float d = dinv[r];
                Cinit[(long long)r * LDC + n] = d * d * val + bn;
            }
        }
    }
}

// ------------------------------ launch ------------------------------

extern "C" void kernel_launch(void* const* d_in, const int* in_sizes, int n_in,
                              void* d_out, int out_size, void* d_ws, size_t ws_size,
                              hipStream_t stream) {
    const float* x  = (const float*)d_in[0];
    const int*   ei = (const int*)d_in[1];     // [2][E]: rows then cols
    const float* W1 = (const float*)d_in[2];
    const float* b1 = (const float*)d_in[3];
    const float* W2 = (const float*)d_in[4];
    const float* b2 = (const float*)d_in[5];
    float* out = (float*)d_out;

    const int N = in_sizes[0] / GCN_F_IN;      // 50000
    const int E = in_sizes[1] / 2;             // 800000
    const int* rows = ei;
    const int* cols = ei + E;

    // carve workspace (256B aligned)
    char* ws = (char*)d_ws;
    size_t off = 0;
    auto carve = [&](size_t bytes) -> void* {
        void* p = ws + off;
        off += (bytes + 255) & ~(size_t)255;
        return p;
    };
    float*    dinv = (float*)carve((size_t)N * 4);
    float*    hlin = (float*)carve((size_t)N * GCN_HID * 4);
    float*    hagg = (float*)carve((size_t)N * GCN_HID * 4);
    float*    t2   = (float*)carve((size_t)N * GCN_NCLS * 4);
    _Float16* w1t  = (_Float16*)carve((size_t)GCN_HID * GCN_F_IN * 2);
    _Float16* w2t  = (_Float16*)carve((size_t)64 * 64 * 2);
    (void)ws_size; (void)n_in; (void)out_size;

    const int T = 256;
    // degree -> dinv (self-loop contributes 1 to every node)
    k_fill_ones<<<(N + T - 1) / T, T, 0, stream>>>(dinv, N);
    k_deg_accum<<<(E + T - 1) / T, T, 0, stream>>>(cols, dinv, E);
    k_inv_sqrt<<<(N + T - 1) / T, T, 0, stream>>>(dinv, N);

    // f16 transposed weights
    k_w1_to_half_t<<<(GCN_HID * GCN_F_IN + T - 1) / T, T, 0, stream>>>(W1, w1t);
    k_w2_to_half_t<<<(64 * 64 + T - 1) / T, T, 0, stream>>>(W2, w2t);

    const int gblk = (N + 15) / 16;

    // layer 1: hlin = x @ W1 (WMMA, epilogue seeds hagg = dinv^2*hlin + b1)
    k_gemm_wmma<GCN_F_IN, GCN_F_IN, GCN_HID, GCN_HID>
        <<<gblk, 128, 0, stream>>>(x, w1t, hlin, hagg, dinv, b1, N);
    {
        long long etot = (long long)E * GCN_HID;
        k_agg_edges<GCN_HID><<<(int)((etot + T - 1) / T), T, 0, stream>>>(rows, cols, dinv, hlin, hagg, E);
        long long tot = (long long)N * GCN_HID;
        k_relu_ip<<<(int)((tot + T - 1) / T), T, 0, stream>>>(hagg, tot);
    }

    // layer 2: t2 = hagg @ W2 (WMMA, epilogue seeds out = dinv^2*t2 + b2)
    k_gemm_wmma<GCN_HID, GCN_HID, GCN_NCLS, GCN_NCLS>
        <<<gblk, 128, 0, stream>>>(hagg, w2t, t2, out, dinv, b2, N);
    {
        long long etot = (long long)E * GCN_NCLS;
        k_agg_edges<GCN_NCLS><<<(int)((etot + T - 1) / T), T, 0, stream>>>(rows, cols, dinv, t2, out, E);
    }
    k_log_softmax_rows<<<(N + 3) / 4, 128, 0, stream>>>(out, N, GCN_NCLS);
}
